// Attention_8409545965959
// MI455X (gfx1250) — matrix operand order repaired and checked by reading, not compiled, not run on hardware
//
#include <hip/hip_runtime.h>
#include <hip/hip_bf16.h>
#include <stdint.h>

// ---------------------------------------------------------------------------
// MHA forward for B=4, N=2048, C=1024, H=16, Dh=64 on gfx1250 (MI455X).
// bf16 WMMA (v_wmma_f32_16x16x32_bf16) everywhere, fp32 accumulate/softmax.
// TDM (tensor_load_to_lds) stages tiles into padded LDS, double-buffered so
// the DMA for step i+1 overlaps the WMMA work of step i.
// ---------------------------------------------------------------------------

typedef __bf16 bf16;
typedef __attribute__((ext_vector_type(16))) __bf16   v16bf;
typedef __attribute__((ext_vector_type(8)))  float    v8f;
typedef __attribute__((ext_vector_type(4)))  uint32_t u32x4;
typedef __attribute__((ext_vector_type(8)))  int      i32x8;
typedef __attribute__((ext_vector_type(4)))  int      i32x4;

union FragAB { v16bf v; u32x4 q[2]; };

__device__ __forceinline__ v8f wmma_bf16(const FragAB& a, const FragAB& b, v8f c) {
  return __builtin_amdgcn_wmma_f32_16x16x32_bf16(false, a.v, false, b.v,
                                                 (short)0, c, false, false);
}

// LDS byte offset of element e in a TDM-padded row: pad of 16B after every 64B.
__device__ __forceinline__ uint32_t eoff(uint32_t e) {
  return e * 2u + (e >> 5) * 16u;
}

#if defined(__has_builtin)
#if __has_builtin(__builtin_amdgcn_tensor_load_to_lds)
#define USE_TDM 1
#endif
#endif

#ifdef USE_TDM
#define TENSOR_WAIT(n) __builtin_amdgcn_s_wait_tensorcnt(n)
#else
#define TENSOR_WAIT(n)
#endif

// Loop-invariant part of the D# (group 1): data_size=2B, pad 4dw per 16dw,
// tensor = rows x cols bf16 with row stride `gstride_elems`.
__device__ __forceinline__ i32x8 tdm_group1(uint32_t cols, uint32_t rows,
                                            uint64_t gstride_elems) {
  i32x8 g1;
  uint32_t d0 = (1u << 16)                     // data_size = 2 bytes
              | (1u << 20)                     // pad_enable
              | (3u << 22)                     // pad_interval = 16 dwords (64B)
              | (3u << 25);                    // pad_amount  = 4 dwords (16B)
  g1[0] = (int)d0;                                                    // wg_mask=0
  g1[1] = (int)((cols & 0xffffu) << 16);                              // dim0 lo
  g1[2] = (int)(((cols >> 16) & 0xffffu) | ((rows & 0xffffu) << 16)); // dim0 hi|dim1 lo
  g1[3] = (int)(((rows >> 16) & 0xffffu) | ((cols & 0xffffu) << 16)); // dim1 hi|tile_dim0
  g1[4] = (int)(rows & 0xffffu);                                      // tile_dim1
  g1[5] = (int)(gstride_elems & 0xffffffffu);                         // stride lo
  g1[6] = (int)((gstride_elems >> 32) & 0xffffu);                     // stride hi
  g1[7] = 0;
  return g1;
}

#ifdef USE_TDM
__device__ __forceinline__ void tdm_issue(uint32_t lds_off, const void* gptr,
                                          const i32x8& g1) {
  uint64_t ga = (uint64_t)(uintptr_t)gptr;
  u32x4 g0;
  g0[0] = 1u;                                  // count=1, user mode, no gather
  g0[1] = lds_off;                             // lds_addr (bytes)
  g0[2] = (uint32_t)(ga & 0xffffffffu);        // global_addr[31:0]
  g0[3] = (uint32_t)((ga >> 32) & 0x01ffffffu) // global_addr[56:32]
          | (2u << 30);                        // type = 2 (image)
  i32x4 z4 = {0, 0, 0, 0};
  i32x8 z8 = {0, 0, 0, 0, 0, 0, 0, 0};
  __builtin_amdgcn_tensor_load_to_lds(g0, g1, z4, z4, z8, 0);
}
#endif

// Issue one tile stage. TDM: wave 0 issues async DMA. Fallback: cooperative
// synchronous copy into the identical padded layout.
__device__ __forceinline__ void stage(char* smem, uint32_t lds_off,
                                      const bf16* g, uint32_t rows,
                                      uint32_t cols, uint32_t gstride,
                                      uint32_t pitch, int tid, int nthr,
                                      int wave, const i32x8& g1) {
#ifdef USE_TDM
  (void)smem; (void)rows; (void)cols; (void)gstride; (void)pitch;
  (void)tid; (void)nthr;
  if (wave == 0) tdm_issue(lds_off, g, g1);
#else
  (void)wave; (void)g1;
  for (uint32_t i = (uint32_t)tid; i < rows * cols; i += (uint32_t)nthr) {
    uint32_t r = i / cols, c = i % cols;
    *(bf16*)(smem + lds_off + r * pitch + eoff(c)) = g[(size_t)r * gstride + c];
  }
#endif
}

// ---------------------------------------------------------------------------
// Problem constants / workspace layout (bytes)
// ---------------------------------------------------------------------------
constexpr int BB = 4, SEQ = 2048, CD = 1024, NH = 16, DH = 64;
constexpr int BN = BB * SEQ;                       // 8192
constexpr size_t WS_XB  = 0;                       // x  bf16        16.78 MB
constexpr size_t WS_WQT = 16777216;                // WqkvT bf16      6.29 MB
constexpr size_t WS_WPT = 23068672;                // WprojT bf16     2.10 MB
constexpr size_t WS_Q   = 25165824;                // Q [B,H,N,Dh]   16.78 MB
constexpr size_t WS_K   = 41943040;                // K [B,H,N,Dh]   16.78 MB
constexpr size_t WS_VT  = 58720256;                // V^T [B,H,Dh,N] 16.78 MB
constexpr size_t WS_CTX = 0;                       // ctx reuses xb region

// ---------------------------------------------------------------------------
// Kernel 1: fp32 -> bf16 (x) and transposed bf16 weights
// ---------------------------------------------------------------------------
__global__ __launch_bounds__(256) void convert_kernel(
    const float* __restrict__ x, const float* __restrict__ wqkv,
    const float* __restrict__ wproj, bf16* __restrict__ xb,
    bf16* __restrict__ wqt, bf16* __restrict__ wpt) {
  const long long NX = (long long)BN * CD;
  const long long NWQ = (long long)CD * 3 * CD;
  const long long NWP = (long long)CD * CD;
  for (long long i = (long long)blockIdx.x * blockDim.x + threadIdx.x;
       i < NX + NWQ + NWP; i += (long long)gridDim.x * blockDim.x) {
    if (i < NX) {
      xb[i] = (bf16)x[i];
    } else if (i < NX + NWQ) {
      long long j = i - NX;
      long long n = j / CD, k = j % CD;          // wqt[n][k] = W[k][n]
      wqt[j] = (bf16)wqkv[k * (3 * CD) + n];
    } else {
      long long j = i - NX - NWQ;
      long long n = j / CD, k = j % CD;
      wpt[j] = (bf16)wproj[k * CD + n];
    }
  }
}

// ---------------------------------------------------------------------------
// GEMM core: block = 256 thr (8 waves), tile 128(M) x 128(N), K step 32,
// double-buffered TDM. LDS: A0 @0, B0 @10240, A1 @20480, B1 @30720 (40960 B).
// Wave w: M offset (w>>1)*32, N offset (w&1)*64 -> 2x4 16x16 accumulators.
// ---------------------------------------------------------------------------
#define GEMM_PITCH 80u
#define GEMM_A_OFF(buf) ((buf) ? 20480u : 0u)
#define GEMM_B_OFF(buf) ((buf) ? 30720u : 10240u)
#define GEMM_LDS 40960u

// ---------------------------------------------------------------------------
// Kernel 2: qkv = xb @ WqkvT^T, scatter into Q / K / V^T (bf16)
// grid: (3C/128, BN/128) = (24, 64)
// ---------------------------------------------------------------------------
__global__ __launch_bounds__(256) void gemm_qkv_kernel(
    const bf16* __restrict__ xb, const bf16* __restrict__ wqt,
    bf16* __restrict__ Q, bf16* __restrict__ K, bf16* __restrict__ Vt) {
  extern __shared__ char smem[];
  const int tid = threadIdx.x, wave = tid >> 5, lane = tid & 31;
  const int ln = lane & 15, half = lane >> 4;
  const int m0 = blockIdx.y * 128, n0 = blockIdx.x * 128;
  const int wm = (wave >> 1) * 32, wn = (wave & 1) * 64;
  const int kA = half ? 16 : 0, kB = half ? 16 : 0;
  const i32x8 g1 = tdm_group1(32, 128, CD);

  // Prologue: stage k-step 0 into buffer 0.
  stage(smem, GEMM_A_OFF(0), xb + (size_t)m0 * CD, 128, 32, CD, GEMM_PITCH,
        tid, 256, wave, g1);
  stage(smem, GEMM_B_OFF(0), wqt + (size_t)n0 * CD, 128, 32, CD, GEMM_PITCH,
        tid, 256, wave, g1);

  v8f acc[2][4] = {};
  for (int it = 0; it < CD / 32; ++it) {
    const int kn = (it + 1) * 32;
    __syncthreads();  // all waves done consuming the buffer we are about to fill
    if (kn < CD) {
      stage(smem, GEMM_A_OFF((it + 1) & 1), xb + (size_t)m0 * CD + kn, 128, 32,
            CD, GEMM_PITCH, tid, 256, wave, g1);
      stage(smem, GEMM_B_OFF((it + 1) & 1), wqt + (size_t)n0 * CD + kn, 128, 32,
            CD, GEMM_PITCH, tid, 256, wave, g1);
      if (wave == 0) { TENSOR_WAIT(2); }  // current buffer's pair has landed
    } else {
      if (wave == 0) { TENSOR_WAIT(0); }
    }
    __syncthreads();  // current buffer visible to all waves

    const uint32_t aoff = GEMM_A_OFF(it & 1), boff = GEMM_B_OFF(it & 1);
    FragAB a[2], b[4];
#pragma unroll
    for (int i = 0; i < 2; i++) {
      const char* rp = smem + aoff + (uint32_t)(wm + i * 16 + ln) * GEMM_PITCH;
      a[i].q[0] = *(const u32x4*)(rp + kA * 2);
      a[i].q[1] = *(const u32x4*)(rp + (kA + 8) * 2);
    }
#pragma unroll
    for (int j = 0; j < 4; j++) {
      const char* rp = smem + boff + (uint32_t)(wn + j * 16 + ln) * GEMM_PITCH;
      b[j].q[0] = *(const u32x4*)(rp + kB * 2);
      b[j].q[1] = *(const u32x4*)(rp + (kB + 8) * 2);
    }
#pragma unroll
    for (int i = 0; i < 2; i++)
#pragma unroll
      for (int j = 0; j < 4; j++)
        acc[i][j] = wmma_bf16(a[i], b[j], acc[i][j]);
  }

  // Scatter: Q/K as [B,H,N,Dh], V transposed as [B,H,Dh,N]
#pragma unroll
  for (int i = 0; i < 2; i++)
#pragma unroll
    for (int j = 0; j < 4; j++)
#pragma unroll
      for (int r = 0; r < 8; r++) {
        int gm = m0 + wm + i * 16 + r + half * 8;   // row in [0, BN)
        int gn = n0 + wn + j * 16 + ln;             // col in [0, 3C)
        int b_ = gm >> 11, nq = gm & (SEQ - 1);
        int which = gn >> 10, c = gn & (CD - 1);
        int h = c >> 6, d = c & (DH - 1);
        bf16 val = (bf16)acc[i][j][r];
        size_t bh = (size_t)(b_ * NH + h);
        if (which == 0)       Q[(bh * SEQ + nq) * DH + d] = val;
        else if (which == 1)  K[(bh * SEQ + nq) * DH + d] = val;
        else                  Vt[(bh * DH + d) * SEQ + nq] = val;
      }
}

// ---------------------------------------------------------------------------
// Kernel 3: flash attention. Block = 128 thr (4 waves), 64 queries/block.
// grid = B*H*(N/64) = 2048. Double-buffered K/V chunks:
// K0 @0, V0 @10240, K1 @20480, V1 @30720 (pitch 160), P @40960 (4x16x144).
// ---------------------------------------------------------------------------
#define FA_K_OFF(buf) ((buf) ? 20480u : 0u)
#define FA_V_OFF(buf) ((buf) ? 30720u : 10240u)
#define FA_P_OFF   40960u
#define FA_KV_PITCH 160u
#define FA_P_PITCH  144u
#define FA_LDS     (40960u + 4u * 16u * 144u)   // 50176

__global__ __launch_bounds__(128) void flash_attn_kernel(
    const bf16* __restrict__ Q, const bf16* __restrict__ Km,
    const bf16* __restrict__ Vt, const int* __restrict__ mask,
    bf16* __restrict__ ctx) {
  extern __shared__ char smem[];
  const int tid = threadIdx.x, wave = tid >> 5, lane = tid & 31;
  const int ln = lane & 15, half = lane >> 4;
  const int bid = blockIdx.x;
  const int qt = bid & 31, h = (bid >> 5) & (NH - 1), b = bid >> 9;
  const int q0 = qt * 64 + wave * 16;
  const size_t bh = (size_t)(b * NH + h);
  const int kA = half ? 16 : 0, kB = half ? 16 : 0;
  const i32x8 g1k = tdm_group1(64, 64, DH);
  const i32x8 g1v = tdm_group1(64, 64, SEQ);
  const bf16* kbase = Km + bh * SEQ * DH;
  const bf16* vbase = Vt + bh * DH * SEQ;

  // Q fragments (Dh = 64 -> two 16x32 A-fragments), loaded once from global.
  FragAB qf[2];
  {
    const bf16* qrow = Q + (bh * SEQ + q0 + ln) * DH;
    qf[0].q[0] = *(const u32x4*)(qrow + kA);
    qf[0].q[1] = *(const u32x4*)(qrow + kA + 8);
    qf[1].q[0] = *(const u32x4*)(qrow + 32 + kA);
    qf[1].q[1] = *(const u32x4*)(qrow + 40 + kA);
  }

  float m[8], l[8];
  v8f o[4] = {};
#pragma unroll
  for (int r = 0; r < 8; r++) { m[r] = -1e30f; l[r] = 0.0f; }
  const int* mrow = mask + b * SEQ;
  const float sc = 0.125f;  // Dh^-0.5

  // Prologue: stage chunk 0 into buffer 0.
  stage(smem, FA_K_OFF(0), kbase, 64, 64, DH, FA_KV_PITCH, tid, 128, wave, g1k);
  stage(smem, FA_V_OFF(0), vbase, 64, 64, SEQ, FA_KV_PITCH, tid, 128, wave, g1v);

  for (int kc = 0; kc < SEQ / 64; ++kc) {
    const int key0 = kc * 64;
    __syncthreads();  // prior chunk's consumers done with the buffer we refill
    if (kc + 1 < SEQ / 64) {
      stage(smem, FA_K_OFF((kc + 1) & 1), kbase + (size_t)(key0 + 64) * DH, 64,
            64, DH, FA_KV_PITCH, tid, 128, wave, g1k);
      stage(smem, FA_V_OFF((kc + 1) & 1), vbase + key0 + 64, 64, 64, SEQ,
            FA_KV_PITCH, tid, 128, wave, g1v);
      if (wave == 0) { TENSOR_WAIT(2); }
    } else {
      if (wave == 0) { TENSOR_WAIT(0); }
    }
    __syncthreads();
    const uint32_t koff = FA_K_OFF(kc & 1), voff = FA_V_OFF(kc & 1);

    // ---- S = Q K^T for 4 key sub-tiles of 16 ----
    v8f s[4];
#pragma unroll
    for (int t = 0; t < 4; t++) {
      const char* rp = smem + koff + (uint32_t)(t * 16 + ln) * FA_KV_PITCH;
      FragAB kf0, kf1;
      kf0.q[0] = *(const u32x4*)(rp + eoff(kB));
      kf0.q[1] = *(const u32x4*)(rp + eoff(kB + 8));
      kf1.q[0] = *(const u32x4*)(rp + eoff(32 + kB));
      kf1.q[1] = *(const u32x4*)(rp + eoff(32 + kB + 8));
      v8f z = {};
      z = wmma_bf16(qf[0], kf0, z);
      z = wmma_bf16(qf[1], kf1, z);
      s[t] = z;
    }

    // ---- scale + key mask (lane's column = key index) ----
#pragma unroll
    for (int t = 0; t < 4; t++) {
      int msk = mrow[key0 + t * 16 + ln] > 0;
#pragma unroll
      for (int r = 0; r < 8; r++)
        s[t][r] = msk ? -10000000.0f : s[t][r] * sc;
    }

    // ---- online softmax (rows live in 16-lane halves) ----
#pragma unroll
    for (int r = 0; r < 8; r++) {
      float tm = fmaxf(fmaxf(s[0][r], s[1][r]), fmaxf(s[2][r], s[3][r]));
      tm = fmaxf(tm, __shfl_xor(tm, 1));
      tm = fmaxf(tm, __shfl_xor(tm, 2));
      tm = fmaxf(tm, __shfl_xor(tm, 4));
      tm = fmaxf(tm, __shfl_xor(tm, 8));
      float mn = fmaxf(m[r], tm);
      float alpha = __expf(m[r] - mn);
      m[r] = mn;
      l[r] = l[r] * alpha;
#pragma unroll
      for (int t2 = 0; t2 < 4; t2++) o[t2][r] = o[t2][r] * alpha;
      float rs = 0.0f;
#pragma unroll
      for (int t = 0; t < 4; t++) {
        float p = __expf(s[t][r] - mn);
        s[t][r] = p;
        rs += p;
      }
      rs += __shfl_xor(rs, 1);
      rs += __shfl_xor(rs, 2);
      rs += __shfl_xor(rs, 4);
      rs += __shfl_xor(rs, 8);
      l[r] += rs;
    }

    // ---- transpose P (C-layout -> A-fragment) through per-wave LDS ----
    const uint32_t pbase = FA_P_OFF + (uint32_t)wave * (16u * FA_P_PITCH);
#pragma unroll
    for (int t = 0; t < 4; t++)
#pragma unroll
      for (int r = 0; r < 8; r++) {
        uint32_t row = (uint32_t)(r + half * 8);
        *(bf16*)(smem + pbase + row * FA_P_PITCH + (uint32_t)(t * 16 + ln) * 2) =
            (bf16)s[t][r];
      }
    __syncthreads();  // orders DS stores before DS loads across lanes

    FragAB pf0, pf1;
    {
      const char* rp = smem + pbase + (uint32_t)ln * FA_P_PITCH;
      pf0.q[0] = *(const u32x4*)(rp + kA * 2);
      pf0.q[1] = *(const u32x4*)(rp + (kA + 8) * 2);
      pf1.q[0] = *(const u32x4*)(rp + (32 + kA) * 2);
      pf1.q[1] = *(const u32x4*)(rp + (40 + kA) * 2);
    }

    // ---- O += P V ----
#pragma unroll
    for (int t2 = 0; t2 < 4; t2++) {
      const char* rp = smem + voff + (uint32_t)(t2 * 16 + ln) * FA_KV_PITCH;
      FragAB vf0, vf1;
      vf0.q[0] = *(const u32x4*)(rp + eoff(kB));
      vf0.q[1] = *(const u32x4*)(rp + eoff(kB + 8));
      vf1.q[0] = *(const u32x4*)(rp + eoff(32 + kB));
      vf1.q[1] = *(const u32x4*)(rp + eoff(32 + kB + 8));
      o[t2] = wmma_bf16(pf0, vf0, o[t2]);
      o[t2] = wmma_bf16(pf1, vf1, o[t2]);
    }
  }

  // ---- epilogue: ctx[b, q, h*64 + d] = O / l ----
#pragma unroll
  for (int t2 = 0; t2 < 4; t2++)
#pragma unroll
    for (int r = 0; r < 8; r++) {
      int q = q0 + r + half * 8;
      int d = t2 * 16 + ln;
      float val = o[t2][r] / l[r];
      ctx[((size_t)b * SEQ + q) * CD + h * DH + d] = (bf16)val;
    }
}

// ---------------------------------------------------------------------------
// Kernel 4: out = ctx @ WprojT^T + bproj  (fp32 output)
// grid: (C/128, BN/128) = (8, 64)
// ---------------------------------------------------------------------------
__global__ __launch_bounds__(256) void gemm_proj_kernel(
    const bf16* __restrict__ ctx, const bf16* __restrict__ wpt,
    const float* __restrict__ bias, float* __restrict__ out) {
  extern __shared__ char smem[];
  const int tid = threadIdx.x, wave = tid >> 5, lane = tid & 31;
  const int ln = lane & 15, half = lane >> 4;
  const int m0 = blockIdx.y * 128, n0 = blockIdx.x * 128;
  const int wm = (wave >> 1) * 32, wn = (wave & 1) * 64;
  const int kA = half ? 16 : 0, kB = half ? 16 : 0;
  const i32x8 g1 = tdm_group1(32, 128, CD);

  stage(smem, GEMM_A_OFF(0), ctx + (size_t)m0 * CD, 128, 32, CD, GEMM_PITCH,
        tid, 256, wave, g1);
  stage(smem, GEMM_B_OFF(0), wpt + (size_t)n0 * CD, 128, 32, CD, GEMM_PITCH,
        tid, 256, wave, g1);

  v8f acc[2][4] = {};
  for (int it = 0; it < CD / 32; ++it) {
    const int kn = (it + 1) * 32;
    __syncthreads();
    if (kn < CD) {
      stage(smem, GEMM_A_OFF((it + 1) & 1), ctx + (size_t)m0 * CD + kn, 128, 32,
            CD, GEMM_PITCH, tid, 256, wave, g1);
      stage(smem, GEMM_B_OFF((it + 1) & 1), wpt + (size_t)n0 * CD + kn, 128, 32,
            CD, GEMM_PITCH, tid, 256, wave, g1);
      if (wave == 0) { TENSOR_WAIT(2); }
    } else {
      if (wave == 0) { TENSOR_WAIT(0); }
    }
    __syncthreads();

    const uint32_t aoff = GEMM_A_OFF(it & 1), boff = GEMM_B_OFF(it & 1);
    FragAB a[2], b[4];
#pragma unroll
    for (int i = 0; i < 2; i++) {
      const char* rp = smem + aoff + (uint32_t)(wm + i * 16 + ln) * GEMM_PITCH;
      a[i].q[0] = *(const u32x4*)(rp + kA * 2);
      a[i].q[1] = *(const u32x4*)(rp + (kA + 8) * 2);
    }
#pragma unroll
    for (int j = 0; j < 4; j++) {
      const char* rp = smem + boff + (uint32_t)(wn + j * 16 + ln) * GEMM_PITCH;
      b[j].q[0] = *(const u32x4*)(rp + kB * 2);
      b[j].q[1] = *(const u32x4*)(rp + (kB + 8) * 2);
    }
#pragma unroll
    for (int i = 0; i < 2; i++)
#pragma unroll
      for (int j = 0; j < 4; j++)
        acc[i][j] = wmma_bf16(a[i], b[j], acc[i][j]);
  }

#pragma unroll
  for (int i = 0; i < 2; i++)
#pragma unroll
    for (int j = 0; j < 4; j++)
#pragma unroll
      for (int r = 0; r < 8; r++) {
        int gm = m0 + wm + i * 16 + r + half * 8;
        int gn = n0 + wn + j * 16 + ln;
        out[(size_t)gm * CD + gn] = acc[i][j][r] + bias[gn];
      }
}

// ---------------------------------------------------------------------------
// Launcher. Workspace requirement: ~75.5 MB.
// ---------------------------------------------------------------------------
extern "C" void kernel_launch(void* const* d_in, const int* in_sizes, int n_in,
                              void* d_out, int out_size, void* d_ws,
                              size_t ws_size, hipStream_t stream) {
  (void)in_sizes; (void)n_in; (void)out_size; (void)ws_size;
  const float* x     = (const float*)d_in[0];
  const float* Wqkv  = (const float*)d_in[1];
  const float* Wproj = (const float*)d_in[2];
  const float* bproj = (const float*)d_in[3];
  const int*   mask  = (const int*)d_in[4];
  float* out = (float*)d_out;
  char* ws = (char*)d_ws;

  bf16* xb  = (bf16*)(ws + WS_XB);
  bf16* wqt = (bf16*)(ws + WS_WQT);
  bf16* wpt = (bf16*)(ws + WS_WPT);
  bf16* Qd  = (bf16*)(ws + WS_Q);
  bf16* Kd  = (bf16*)(ws + WS_K);
  bf16* Vtd = (bf16*)(ws + WS_VT);
  bf16* ctx = (bf16*)(ws + WS_CTX);   // reuses xb region (xb dead by then)

  convert_kernel<<<4096, 256, 0, stream>>>(x, Wqkv, Wproj, xb, wqt, wpt);
  gemm_qkv_kernel<<<dim3(24, 64), 256, GEMM_LDS, stream>>>(xb, wqt, Qd, Kd, Vtd);
  flash_attn_kernel<<<BB * NH * (SEQ / 64), 128, FA_LDS, stream>>>(Qd, Kd, Vtd,
                                                                   mask, ctx);
  gemm_proj_kernel<<<dim3(8, 64), 256, GEMM_LDS, stream>>>(ctx, wpt, bproj, out);
}
